// GraphTransformer_13975823581434
// MI455X (gfx1250) — matrix operand order repaired and checked by reading, not compiled, not run on hardware
//
#include <hip/hip_runtime.h>
#include <hip/hip_bf16.h>
#include <math.h>

// ---------------------------------------------------------------------------
// GraphTransformer (2x TransformerConv H=2,HID=64 + linear) for gfx1250.
// GEMMs via v_wmma_f32_16x16x32_bf16; edge phase fp32 gather/scatter (L2-
// resident: qkvs table ~205MB, q/k/v slices fit 192MB L2).
// ---------------------------------------------------------------------------

typedef __attribute__((ext_vector_type(16))) __bf16 v16bf;
typedef __attribute__((ext_vector_type(8)))  __bf16 bf16x8;
typedef __attribute__((ext_vector_type(8)))  float  v8f;

// monotonic uint encoding of float for atomicMax-based segment max
__device__ __forceinline__ unsigned f32_key(float f) {
  unsigned u = __float_as_uint(f);
  return (u & 0x80000000u) ? ~u : (u | 0x80000000u);
}
__device__ __forceinline__ float key_f32(unsigned k) {
  unsigned u = (k & 0x80000000u) ? (k & 0x7fffffffu) : ~k;
  return __uint_as_float(u);
}

// ---------------------------------------------------------------------------
__global__ void zero_f32(float* __restrict__ p, long long n) {
  long long i = (long long)blockIdx.x * blockDim.x + threadIdx.x;
  long long stride = (long long)gridDim.x * blockDim.x;
  for (; i < n; i += stride) p[i] = 0.0f;
}

__global__ void cvt_f32_bf16(const float* __restrict__ src,
                             __bf16* __restrict__ dst, long long n) {
  long long i = (long long)blockIdx.x * blockDim.x + threadIdx.x;
  if (i < n) dst[i] = (__bf16)src[i];
}

// W [128, cols] row-major (fan_in x fan_out)  ->  Bt [128][128] bf16 where
// Bt[nCol][k] = W[k][nCol]; columns >= cols zero-padded.
__global__ void prep_weight(const float* __restrict__ W, int cols,
                            __bf16* __restrict__ Bt) {
  int idx = blockIdx.x * blockDim.x + threadIdx.x;  // 128*128 threads
  if (idx >= 128 * 128) return;
  int ncol = idx >> 7;
  int k    = idx & 127;
  float v = (ncol < cols) ? W[k * cols + ncol] : 0.0f;
  Bt[idx] = (__bf16)v;
}

// ---------------------------------------------------------------------------
// C[m0+16 x 64-col group] = A[M,128](bf16) x Bt^T + bias, via WMMA bf16.
// Block = 128 threads = 4 waves; wave w owns the 16x16 tile at n0 = gy*64+w*16.
// A-frag lane layout (16-bit A 16x32): lane = half*16 + row; elems 0..7 are
// K = kc+half*8+0..7, elems 8..15 are K = kc+16+half*8+0..7  (two b128 loads).
// B-frag (16-bit B 32x16): lane = half*16 + col; elems 0..7 -> K = kc+half*8+i, 8..15 -> K = kc+16+half*8+i-8
// (same K order as A; two b128 loads from the pre-transposed Bt).
// C/D: VGPR r -> row m0 + r + 8*half, col n0 + (lane&15).
// ---------------------------------------------------------------------------
__global__ void __launch_bounds__(128)
gemm_bf16_wmma(const __bf16* __restrict__ A, const __bf16* __restrict__ Bt,
               const float* __restrict__ bias, float* __restrict__ C,
               int ldC, int nreal) {
  const int wave = threadIdx.x >> 5;
  const int lane = threadIdx.x & 31;
  const int half = lane >> 4;
  const int l16  = lane & 15;
  const int m0 = blockIdx.x * 16;
  const int n0 = blockIdx.y * 64 + wave * 16;

  const __bf16* ap = A  + (long long)(m0 + l16) * 128 + half * 8;
  const __bf16* bp = Bt + (long long)(n0 + l16) * 128 + half * 8;

  v8f acc = {};
#pragma unroll
  for (int kc = 0; kc < 128; kc += 32) {
    if (kc < 96) __builtin_prefetch(ap + kc + 32, 0, 1);  // global_prefetch_b8
    bf16x8 alo = *(const bf16x8*)(ap + kc);
    bf16x8 ahi = *(const bf16x8*)(ap + kc + 16);
    bf16x8 blo = *(const bf16x8*)(bp + kc);
    bf16x8 bhi = *(const bf16x8*)(bp + kc + 16);
    v16bf avec, bvec;
#pragma unroll
    for (int i = 0; i < 8; ++i) {
      avec[i] = alo[i]; avec[i + 8] = ahi[i];
      bvec[i] = blo[i]; bvec[i + 8] = bhi[i];
    }
    acc = __builtin_amdgcn_wmma_f32_16x16x32_bf16(
        false, avec, false, bvec, (short)0, acc, false, false);
  }

  const int col = n0 + l16;
  if (col < nreal) {
    float b = bias[col];
#pragma unroll
    for (int r = 0; r < 8; ++r) {
      int row = m0 + r + half * 8;
      C[(long long)row * ldC + col] = acc[r] + b;
    }
  }
}

// ---------------------------------------------------------------------------
// Edge phase. QKVS layout per node: [q(0:128) | k(128:256) | v(256:384) | s(384:512)]
// ---------------------------------------------------------------------------
__global__ void edge_alpha(const long long* __restrict__ ei, long long E,
                           const float* __restrict__ QKVS,
                           float* __restrict__ ALPHA,
                           unsigned* __restrict__ MKEY) {
  long long e = (long long)blockIdx.x * blockDim.x + threadIdx.x;
  if (e >= E) return;
  int src = (int)ei[e];
  int dst = (int)ei[E + e];
  const float4* q = (const float4*)(QKVS + (long long)dst * 512);
  const float4* k = (const float4*)(QKVS + (long long)src * 512 + 128);
  float a0 = 0.f, a1 = 0.f;
#pragma unroll
  for (int i = 0; i < 16; ++i) {
    float4 qa = q[i], ka = k[i];
    a0 += qa.x * ka.x + qa.y * ka.y + qa.z * ka.z + qa.w * ka.w;
  }
#pragma unroll
  for (int i = 16; i < 32; ++i) {
    float4 qa = q[i], ka = k[i];
    a1 += qa.x * ka.x + qa.y * ka.y + qa.z * ka.z + qa.w * ka.w;
  }
  a0 *= 0.125f;  // 1/sqrt(64)
  a1 *= 0.125f;
  ALPHA[2 * e]     = a0;
  ALPHA[2 * e + 1] = a1;
  atomicMax(MKEY + 2 * (long long)dst,     f32_key(a0));
  atomicMax(MKEY + 2 * (long long)dst + 1, f32_key(a1));
}

__global__ void edge_exp(const long long* __restrict__ ei, long long E,
                         float* __restrict__ ALPHA,
                         const unsigned* __restrict__ MKEY,
                         float* __restrict__ DEN) {
  long long e = (long long)blockIdx.x * blockDim.x + threadIdx.x;
  if (e >= E) return;
  int dst = (int)ei[E + e];
  float m0 = key_f32(MKEY[2 * (long long)dst]);
  float m1 = key_f32(MKEY[2 * (long long)dst + 1]);
  float a0 = __expf(ALPHA[2 * e] - m0);
  float a1 = __expf(ALPHA[2 * e + 1] - m1);
  ALPHA[2 * e]     = a0;
  ALPHA[2 * e + 1] = a1;
  atomicAdd(DEN + 2 * (long long)dst,     a0);
  atomicAdd(DEN + 2 * (long long)dst + 1, a1);
}

// one thread = (edge, 4 channels): 32 threads per edge cover 128 channels.
__global__ void edge_scatter(const long long* __restrict__ ei, long long E,
                             const float* __restrict__ ALPHA,
                             const float* __restrict__ QKVS,
                             float* __restrict__ AGG) {
  long long tid = (long long)blockIdx.x * blockDim.x + threadIdx.x;
  if (tid >= E * 32) return;
  long long e = tid >> 5;
  int c0 = (int)(tid & 31) * 4;
  int src = (int)ei[e];
  int dst = (int)ei[E + e];
  float a = ALPHA[2 * e + (c0 >> 6)];
  float4 v = *(const float4*)(QKVS + (long long)src * 512 + 256 + c0);
  float* o = AGG + (long long)dst * 128 + c0;
  atomicAdd(o + 0, v.x * a);
  atomicAdd(o + 1, v.y * a);
  atomicAdd(o + 2, v.z * a);
  atomicAdd(o + 3, v.w * a);
}

// h = AGG/(den+eps) + skip; optional ELU; emit bf16 for the next GEMM.
__global__ void combine(const float* __restrict__ AGG,
                        const float* __restrict__ DEN,
                        const float* __restrict__ QKVS,
                        __bf16* __restrict__ OUT16,
                        long long total, int elu) {
  long long idx = (long long)blockIdx.x * blockDim.x + threadIdx.x;
  if (idx >= total) return;
  long long n = idx >> 7;
  int c = (int)(idx & 127);
  int h = c >> 6;
  float val = AGG[idx] / (DEN[2 * n + h] + 1e-16f) + QKVS[n * 512 + 384 + c];
  if (elu) val = (val > 0.f) ? val : (__expf(val) - 1.f);
  OUT16[idx] = (__bf16)val;
}

// ---------------------------------------------------------------------------
extern "C" void kernel_launch(void* const* d_in, const int* in_sizes, int n_in,
                              void* d_out, int out_size, void* d_ws, size_t ws_size,
                              hipStream_t stream) {
  (void)n_in; (void)out_size; (void)ws_size;
  const float*     x  = (const float*)d_in[0];
  const long long* ei = (const long long*)d_in[1];  // int64 [2, E]
  const float* W[9];
  const float* B[9];
  for (int i = 0; i < 8; ++i) {            // q1,k1,v1,s1,q2,k2,v2,s2
    W[i] = (const float*)d_in[2 + 2 * i];
    B[i] = (const float*)d_in[3 + 2 * i];
  }
  W[8] = (const float*)d_in[18];           // Wlin
  B[8] = (const float*)d_in[19];           // blin

  const long long N = (long long)in_sizes[0] / 128;
  const long long E = (long long)in_sizes[1] / 2;

  // ---- workspace layout (256B aligned slabs) ----
  char* ws = (char*)d_ws;
  size_t off = 0;
  auto take = [&](size_t bytes) -> char* {
    char* p = ws + off;
    off += (bytes + 255) & ~(size_t)255;
    return p;
  };
  __bf16*   F16   = (__bf16*)  take((size_t)N * 128 * 2);      // bf16 features
  __bf16*   WT    = (__bf16*)  take((size_t)9 * 128 * 128 * 2);// transposed bf16 weights
  float*    QKVS  = (float*)   take((size_t)N * 512 * 4);      // q|k|v|s per node
  float*    ALPHA = (float*)   take((size_t)E * 2 * 4);        // per-edge per-head
  unsigned* MKEY  = (unsigned*)take((size_t)N * 2 * 4);        // segment max (encoded)
  float*    DEN   = (float*)   take((size_t)N * 2 * 4);        // softmax denom
  float*    AGG   = (float*)   take((size_t)N * 128 * 4);      // scatter accumulator

  const int TB = 256;
  dim3 blk(TB);
  const long long NF = N * 128;

  // ---- prep: bf16 features + transposed bf16 weights ----
  cvt_f32_bf16<<<dim3((unsigned)((NF + TB - 1) / TB)), blk, 0, stream>>>(x, F16, NF);
  for (int w = 0; w < 9; ++w) {
    prep_weight<<<dim3((128 * 128 + TB - 1) / TB), blk, 0, stream>>>(
        W[w], (w == 8) ? 40 : 128, WT + (size_t)w * 128 * 128);
  }

  const unsigned mtiles = (unsigned)(N / 16);  // 100000/16 = 6250 exact

  for (int layer = 0; layer < 2; ++layer) {
    const int wb = layer * 4;  // weight base: 0 (layer1) or 4 (layer2)
    // q,k,v,s projections -> QKVS slots 0/128/256/384, ldC = 512
    for (int s = 0; s < 4; ++s) {
      gemm_bf16_wmma<<<dim3(mtiles, 2), dim3(128), 0, stream>>>(
          F16, WT + (size_t)(wb + s) * 128 * 128, B[wb + s],
          QKVS + s * 128, 512, 128);
    }
    // reset per-layer accumulators
    zero_f32<<<dim3(2048), blk, 0, stream>>>(AGG, NF);
    zero_f32<<<dim3(256), blk, 0, stream>>>((float*)MKEY, N * 2);
    zero_f32<<<dim3(256), blk, 0, stream>>>(DEN, N * 2);
    // edge phase
    edge_alpha<<<dim3((unsigned)((E + TB - 1) / TB)), blk, 0, stream>>>(
        ei, E, QKVS, ALPHA, MKEY);
    edge_exp<<<dim3((unsigned)((E + TB - 1) / TB)), blk, 0, stream>>>(
        ei, E, ALPHA, MKEY, DEN);
    edge_scatter<<<dim3((unsigned)((E * 32 + TB - 1) / TB)), blk, 0, stream>>>(
        ei, E, ALPHA, QKVS, AGG);
    // combine + (ELU for layer 1); writes bf16 features for next GEMM
    combine<<<dim3((unsigned)((NF + TB - 1) / TB)), blk, 0, stream>>>(
        AGG, DEN, QKVS, F16, NF, layer == 0 ? 1 : 0);
  }

  // ---- final classifier: [N,128] @ [128,40] (+blin), N-tiles padded to 64 ----
  gemm_bf16_wmma<<<dim3(mtiles, 1), dim3(128), 0, stream>>>(
      F16, WT + (size_t)8 * 128 * 128, B[8], (float*)d_out, 40, 40);
}
